// E_GCL_317827580691
// MI455X (gfx1250) — matrix operand order repaired and checked by reading, not compiled, not run on hardware
//
#include <hip/hip_runtime.h>
#include <hip/hip_bf16.h>
#include <math.h>

typedef __attribute__((ext_vector_type(16))) _Float16 v16h;
typedef __attribute__((ext_vector_type(8)))  float    v8f;

#define NN 1024
#define HH 64
#define EW1_STRIDE (2*HH + 2)   // 130
#define COORDS_RANGE 10.0f

// fast rcp (v_rcp_f32, ~1ulp) instead of IEEE division chains
__device__ __forceinline__ float fast_rcp(float v) {
  return __builtin_amdgcn_rcpf(v);
}
__device__ __forceinline__ float silu_f(float v) {
  return v * fast_rcp(1.0f + __expf(-v));
}
__device__ __forceinline__ float sigmoid_f(float v) {
  return fast_rcp(1.0f + __expf(-v));
}
__device__ __forceinline__ float tanh_fast(float v) {
  // saturates to +/-1 for large |v| (exp->inf => rcp->0 ; exp->0 => 1-2*1= -1)
  return 1.0f - 2.0f * fast_rcp(__expf(2.0f * v) + 1.0f);
}

// ---------------------------------------------------------------------------
// Kernel 1: hA = h @ Wa^T, hB = h @ Wb^T, hN = h @ nw1[:, :H]^T
// ---------------------------------------------------------------------------
__global__ void __launch_bounds__(64) precompute_kernel(
    const float* __restrict__ h, const float* __restrict__ ew1,
    const float* __restrict__ nw1,
    float* __restrict__ hA, float* __restrict__ hB, float* __restrict__ hN)
{
  __shared__ float hr[HH];
  const int i = blockIdx.x, o = threadIdx.x;
  hr[o] = h[i * HH + o];
  __syncthreads();
  float a = 0.f, b = 0.f, c = 0.f;
#pragma unroll 8
  for (int k = 0; k < HH; ++k) {
    const float hk = hr[k];
    a += hk * ew1[o * EW1_STRIDE + k];
    b += hk * ew1[o * EW1_STRIDE + HH + k];
    c += hk * nw1[o * (2 * HH) + k];
  }
  hA[i * HH + o] = a;
  hB[i * HH + o] = b;
  hN[i * HH + o] = c;
}

// ---------------------------------------------------------------------------
// WMMA fragment loaders (wave32, 16x16x32 f16)
// ---------------------------------------------------------------------------
__device__ __forceinline__ v16h load_fragA(const _Float16* buf, int row0, int lane, int ks) {
  const int m = lane & 15, half = lane >> 4;
  const _Float16* p = buf + (row0 + m) * HH;
  v16h a;
#pragma unroll
  for (int v = 0; v < 8; ++v) {
    const int k = ((v & 3) << 1) + ((v >> 2) << 4) + (half << 3) + ks;
    a[2 * v]     = p[k];
    a[2 * v + 1] = p[k + 1];
  }
  return a;
}

__device__ __forceinline__ v16h load_fragB(const _Float16* buf, int col0, int lane, int ks) {
  const int n = lane & 15, half = lane >> 4;
  const _Float16* p = buf + (col0 + n) * HH;
  v16h b;
#pragma unroll
  for (int v = 0; v < 8; ++v) {
    const int k = ((v & 3) << 1) + ((v >> 2) << 4) + (half << 3) + ks;
    b[2 * v]     = p[k];
    b[2 * v + 1] = p[k + 1];
  }
  return b;
}

// ---------------------------------------------------------------------------
// Kernel 2: fused edge pipeline over a 16(i) x 16(j) tile = 256 edges.
// ---------------------------------------------------------------------------
__global__ void __launch_bounds__(256) edge_kernel(
    const float* __restrict__ x, const float* __restrict__ Lp,
    const float* __restrict__ ew1, const float* __restrict__ eb1,
    const float* __restrict__ ew2, const float* __restrict__ eb2,
    const float* __restrict__ chw, const float* __restrict__ chb,
    const float* __restrict__ cww, const float* __restrict__ aw,
    const float* __restrict__ ab,
    const float* __restrict__ hA, const float* __restrict__ hB,
    float* __restrict__ agg, float* __restrict__ delta)
{
  __shared__ _Float16 Abuf[256 * HH];   // 32 KB: silu(pre), then edge*att (in place)
  __shared__ _Float16 B1[HH * HH];      //  8 KB: ew2 [n][k]
  __shared__ _Float16 B2[HH * HH];      //  8 KB: chw [n][k]
  __shared__ float hAs[16 * HH];        //  4 KB
  __shared__ float hBs[16 * HH];        //  4 KB
  __shared__ float radial_s[256];       //  1 KB
  __shared__ float coord_s[256 * 3];    //  3 KB

  const int tid  = threadIdx.x;
  const int lane = tid & 31;
  const int wave = tid >> 5;
  const int i0 = blockIdx.x * 16;
  const int j0 = blockIdx.y * 16;
  const float L = Lp[0];

  // ---- phase 1: geometry + staging -------------------------------------
  {
    const int e = tid, ii = e >> 4, jj = e & 15;
    float r2 = 0.f;
    float df[3];
#pragma unroll
    for (int d = 0; d < 3; ++d) {
      const float dx = x[(i0 + ii) * 3 + d] - x[(j0 + jj) * 3 + d];
      float t = dx + 0.5f * L;
      t = t - floorf(t / L) * L;       // jnp.remainder (precise div)
      const float dd = t - 0.5f * L;
      df[d] = dd;
      r2 += dd * dd;
    }
    radial_s[e] = r2;
    const float inv = fast_rcp(sqrtf(r2 + 1e-8f) + 1.0f);
#pragma unroll
    for (int d = 0; d < 3; ++d) coord_s[e * 3 + d] = df[d] * inv;
  }
  for (int idx = tid; idx < 16 * HH; idx += 256) {
    hAs[idx] = hA[i0 * HH + idx];
    hBs[idx] = hB[j0 * HH + idx];
  }
  for (int idx = tid; idx < HH * HH; idx += 256) {
    B1[idx] = (_Float16)ew2[idx];
    B2[idx] = (_Float16)chw[idx];
  }
  __syncthreads();

  // ---- phase 2: pre = hA[i] + hB[j] + radial*wr + eb1 ; silu -> Abuf ----
  {
    const int k = tid & 63;
    const int ebase = tid >> 6;
    const float wr_k  = ew1[k * EW1_STRIDE + 2 * HH] + ew1[k * EW1_STRIDE + 2 * HH + 1];
    const float eb1_k = eb1[k];
    for (int e = ebase; e < 256; e += 4) {
      const int ii = e >> 4, jj = e & 15;
      const float pre = hAs[ii * HH + k] + hBs[jj * HH + k] + radial_s[e] * wr_k + eb1_k;
      Abuf[e * HH + k] = (_Float16)silu_f(pre);
    }
  }
  __syncthreads();

  // ---- phase 3: per-wave WMMA pipeline (M-tiles 2w, 2w+1) ---------------
  const int ln   = lane & 15;
  const int half = lane >> 4;
  float aw_l[4], cww_l[4], eb2_l[4], chb_l[4];
#pragma unroll
  for (int nt = 0; nt < 4; ++nt) {
    aw_l[nt]  = aw[nt * 16 + ln];
    cww_l[nt] = cww[nt * 16 + ln];
    eb2_l[nt] = eb2[nt * 16 + ln];
    chb_l[nt] = chb[nt * 16 + ln];
  }
  const float ab0 = ab[0];

#pragma unroll
  for (int tt = 0; tt < 2; ++tt) {
    const int t = 2 * wave + tt;
    const int row0 = t * 16;

    // GEMM1: edge_pre = silu(pre) @ ew2^T  (f32 accum)
    v8f acc[4];
#pragma unroll
    for (int nt = 0; nt < 4; ++nt) acc[nt] = (v8f){};
#pragma unroll
    for (int ks = 0; ks < HH; ks += 32) {
      const v16h a = load_fragA(Abuf, row0, lane, ks);
#pragma unroll
      for (int nt = 0; nt < 4; ++nt) {
        const v16h b = load_fragB(B1, nt * 16, lane, ks);
        acc[nt] = __builtin_amdgcn_wmma_f32_16x16x32_f16(
            false, a, false, b, (short)0, acc[nt], false, false);
      }
    }
    // bias + silu
#pragma unroll
    for (int nt = 0; nt < 4; ++nt)
#pragma unroll
      for (int r = 0; r < 8; ++r)
        acc[nt][r] = silu_f(acc[nt][r] + eb2_l[nt]);

    // attention: row sums via cross-lane butterflies, then scale
#pragma unroll
    for (int r = 0; r < 8; ++r) {
      float s = acc[0][r] * aw_l[0] + acc[1][r] * aw_l[1] +
                acc[2][r] * aw_l[2] + acc[3][r] * aw_l[3];
      s += __shfl_xor(s, 1, 32);
      s += __shfl_xor(s, 2, 32);
      s += __shfl_xor(s, 4, 32);
      s += __shfl_xor(s, 8, 32);
      const float att = sigmoid_f(s + ab0);
#pragma unroll
      for (int nt = 0; nt < 4; ++nt) acc[nt][r] *= att;
    }

    // write scaled edge back to our own LDS rows (A for GEMM2) + agg atomics
#pragma unroll
    for (int nt = 0; nt < 4; ++nt) {
      float colsum = 0.f;
#pragma unroll
      for (int r = 0; r < 8; ++r) {
        Abuf[(row0 + r + 8 * half) * HH + nt * 16 + ln] = (_Float16)acc[nt][r];
        colsum += acc[nt][r];
      }
      colsum += __shfl_xor(colsum, 16, 32);     // sum over all 16 j rows
      if (half == 0)
        atomicAdd(&agg[(i0 + t) * HH + nt * 16 + ln], colsum);
    }
    asm volatile("s_wait_dscnt 0" ::: "memory");  // LDS RAW: edge write -> fragA read

    // GEMM2: ch_pre = edge @ chw^T
    v8f acc2[4];
#pragma unroll
    for (int nt = 0; nt < 4; ++nt) acc2[nt] = (v8f){};
#pragma unroll
    for (int ks = 0; ks < HH; ks += 32) {
      const v16h a = load_fragA(Abuf, row0, lane, ks);
#pragma unroll
      for (int nt = 0; nt < 4; ++nt) {
        const v16h b = load_fragB(B2, nt * 16, lane, ks);
        acc2[nt] = __builtin_amdgcn_wmma_f32_16x16x32_f16(
            false, a, false, b, (short)0, acc2[nt], false, false);
      }
    }

    // ch = silu(+chb); w = tanh(ch . cww) * 10; delta accumulation
    float pd0 = 0.f, pd1 = 0.f, pd2 = 0.f;
#pragma unroll
    for (int r = 0; r < 8; ++r) {
      float s = 0.f;
#pragma unroll
      for (int nt = 0; nt < 4; ++nt) {
        const float chv = silu_f(acc2[nt][r] + chb_l[nt]);
        s += chv * cww_l[nt];
      }
      s += __shfl_xor(s, 1, 32);
      s += __shfl_xor(s, 2, 32);
      s += __shfl_xor(s, 4, 32);
      s += __shfl_xor(s, 8, 32);
      const float wrow = tanh_fast(s) * COORDS_RANGE;
      const int e = row0 + r + 8 * half;      // jj = r + 8*half
      pd0 += wrow * coord_s[e * 3 + 0];
      pd1 += wrow * coord_s[e * 3 + 1];
      pd2 += wrow * coord_s[e * 3 + 2];
    }
    pd0 += __shfl_xor(pd0, 16, 32);
    pd1 += __shfl_xor(pd1, 16, 32);
    pd2 += __shfl_xor(pd2, 16, 32);
    if (lane == 0) {
      atomicAdd(&delta[(i0 + t) * 3 + 0], pd0);
      atomicAdd(&delta[(i0 + t) * 3 + 1], pd1);
      atomicAdd(&delta[(i0 + t) * 3 + 2], pd2);
    }
  }
}

// ---------------------------------------------------------------------------
// Kernel 3: node update + coordinate wrap.
// ---------------------------------------------------------------------------
__global__ void __launch_bounds__(64) node_kernel(
    const float* __restrict__ h, const float* __restrict__ x,
    const float* __restrict__ Lp,
    const float* __restrict__ nw1, const float* __restrict__ nb1,
    const float* __restrict__ nw2, const float* __restrict__ nb2,
    const float* __restrict__ hN, const float* __restrict__ agg,
    const float* __restrict__ delta,
    float* __restrict__ h_out, float* __restrict__ x_out)
{
  __shared__ float s[HH], aggr[HH], hr[HH];
  const int i = blockIdx.x, o = threadIdx.x;
  aggr[o] = agg[i * HH + o];
  hr[o]   = h[i * HH + o];
  __syncthreads();
  float p = hN[i * HH + o] + nb1[o];
#pragma unroll 8
  for (int k = 0; k < HH; ++k)
    p += aggr[k] * nw1[o * (2 * HH) + HH + k];
  s[o] = silu_f(p);
  __syncthreads();
  float out = hr[o] + nb2[o];
#pragma unroll 8
  for (int k = 0; k < HH; ++k)
    out += s[k] * nw2[o * HH + k];
  h_out[i * HH + o] = out;
  if (o < 3) {
    const float L = Lp[0];
    const float v = x[i * 3 + o] + delta[i * 3 + o];
    x_out[i * 3 + o] = v - floorf(v / L) * L;  // jnp.remainder (precise div)
  }
}

// ---------------------------------------------------------------------------
extern "C" void kernel_launch(void* const* d_in, const int* in_sizes, int n_in,
                              void* d_out, int out_size, void* d_ws, size_t ws_size,
                              hipStream_t stream) {
  const float* h   = (const float*)d_in[0];
  const float* x   = (const float*)d_in[1];
  const float* L   = (const float*)d_in[2];
  const float* ew1 = (const float*)d_in[3];
  const float* eb1 = (const float*)d_in[4];
  const float* ew2 = (const float*)d_in[5];
  const float* eb2 = (const float*)d_in[6];
  const float* nw1 = (const float*)d_in[7];
  const float* nb1 = (const float*)d_in[8];
  const float* nw2 = (const float*)d_in[9];
  const float* nb2 = (const float*)d_in[10];
  const float* chw = (const float*)d_in[11];
  const float* chb = (const float*)d_in[12];
  const float* cww = (const float*)d_in[13];
  const float* aw  = (const float*)d_in[14];
  const float* ab  = (const float*)d_in[15];

  float* ws    = (float*)d_ws;
  float* hA    = ws;                    // N*H
  float* hB    = ws + 1 * NN * HH;      // N*H
  float* hN    = ws + 2 * NN * HH;      // N*H
  float* agg   = ws + 3 * NN * HH;      // N*H (atomics)
  float* delta = ws + 4 * NN * HH;      // N*3 (atomics)

  hipMemsetAsync(agg, 0, (NN * HH + NN * 3) * sizeof(float), stream);

  precompute_kernel<<<NN, HH, 0, stream>>>(h, ew1, nw1, hA, hB, hN);

  edge_kernel<<<dim3(NN / 16, NN / 16), 256, 0, stream>>>(
      x, L, ew1, eb1, ew2, eb2, chw, chb, cww, aw, ab, hA, hB, agg, delta);

  node_kernel<<<NN, HH, 0, stream>>>(
      h, x, L, nw1, nb1, nw2, nb2, hN, agg, delta,
      (float*)d_out, (float*)d_out + NN * HH);
}
